// MovingAverageGatedAttention_45638322487506
// MI455X (gfx1250) — matrix operand order repaired and checked by reading, not compiled, not run on hardware
//
#include <hip/hip_runtime.h>
#include <hip/hip_bf16.h>

// ---------------------------------------------------------------------------
// MEGA block for MI455X (gfx1250, wave32, WMMA).
// GEMMs + attention: v_wmma_f32_16x16x32_bf16, fp32 accumulation.
// All WMMA operands are bf16 in global memory: weights transpose-converted
// once per launch (Nt x K), activations written bf16 by producers, V written
// transposed by the v-GEMM. Every LDS fragment / staging access is a
// contiguous 16B run -> ds_load_b128 / ds_store_b128 / global_load_b128.
// GEMM: 128x128 block tile, BK=32, 8 waves, double-buffered LDS; all 8
// B-fragments preloaded into distinct registers so WMMAs issue back-to-back
// under partial s_wait_dscnt instead of one full LDS round-trip per WMMA.
// ---------------------------------------------------------------------------

typedef __attribute__((ext_vector_type(16))) __bf16        v16bf;
typedef __attribute__((ext_vector_type(8)))  float         v8f;
typedef __attribute__((ext_vector_type(4)))  unsigned int  u32x4;

union bfrag16 { v16bf v; unsigned short u[16]; u32x4 q[2]; };
union pack8   { u32x4 q;    unsigned short u[8];  };

// Problem constants
constexpr int Bc  = 2;
constexpr int Sc  = 2048;
constexpr int Dc  = 2048;
constexpr int Zc  = 512;
constexpr int HDc = 4096;
constexpr int Hc  = 8;
constexpr int Gc  = 32;   // GROUPS; dg = 64, ZH = 64, VH = 512, N(ema) = 8

__device__ __forceinline__ unsigned short f32_to_bf16(float f) {
  union { float f; unsigned int u; } c; c.f = f;
  unsigned int r = c.u + 0x7FFFu + ((c.u >> 16) & 1u);  // RNE
  return (unsigned short)(r >> 16);
}
__device__ __forceinline__ float sigmoidf_(float x) { return 1.f / (1.f + __expf(-x)); }

// ---------------------------------------------------------------------------
// 0) tiled transpose + f32->bf16: in (K x N) f32  ->  out (N x K) bf16
// ---------------------------------------------------------------------------
__global__ __launch_bounds__(256) void cvt_t(const float* __restrict__ in,
                                             unsigned short* __restrict__ out,
                                             int K, int N) {
  __shared__ unsigned short tile[32][33];
  const int nb = blockIdx.x * 32, kb = blockIdx.y * 32;
  const int tx = threadIdx.x, ty = threadIdx.y;      // 32 x 8
  for (int i = 0; i < 32; i += 8)
    tile[ty + i][tx] = f32_to_bf16(in[(size_t)(kb + ty + i) * N + nb + tx]);
  __syncthreads();
  for (int i = 0; i < 32; i += 8)
    out[(size_t)(nb + ty + i) * K + kb + tx] = tile[tx][ty + i];
}

// ---------------------------------------------------------------------------
// 1) timestep_norm partial sums per (row, group)
// ---------------------------------------------------------------------------
__global__ __launch_bounds__(256) void tsn_partial(const float* __restrict__ x,
                                                   float* __restrict__ s1p,
                                                   float* __restrict__ s2p) {
  const int row = blockIdx.x, t = threadIdx.x;
  const float* xp = x + (size_t)row * Dc + t * 8;
  float a = 0.f, q = 0.f;
  for (int i = 0; i < 8; ++i) { float v = xp[i]; a += v; q += v * v; }
  for (int m = 4; m >= 1; m >>= 1) { a += __shfl_xor(a, m, 8); q += __shfl_xor(q, m, 8); }
  if ((t & 7) == 0) {
    int g = t >> 3;
    s1p[(size_t)row * Gc + g] = a;
    s2p[(size_t)row * Gc + g] = q;
  }
}

// 2) cumulative scan over S per (b,g)
__global__ void tsn_scan(const float* __restrict__ s1p, const float* __restrict__ s2p,
                         float* __restrict__ meanp, float* __restrict__ rstdp) {
  const int t = threadIdx.x;
  if (t >= Bc * Gc) return;
  const int b = t >> 5, g = t & 31;
  float a1 = 0.f, a2 = 0.f;
  for (int s = 0; s < Sc; ++s) {
    size_t idx = ((size_t)(b * Sc + s)) * Gc + g;
    a1 += s1p[idx]; a2 += s2p[idx];
    float cnt  = (float)(s + 1) * 64.f;
    float mean = a1 / cnt;
    float var  = a2 / cnt - mean * mean;
    meanp[idx] = mean;
    rstdp[idx] = rsqrtf(var + 1e-5f);
  }
}

// 3) normalize; writes fp32 (for EMA) and bf16 (GEMM A operand)
__global__ __launch_bounds__(256) void tsn_norm(const float* __restrict__ x,
                                                const float* __restrict__ meanp,
                                                const float* __restrict__ rstdp,
                                                const float* __restrict__ w,
                                                const float* __restrict__ bb,
                                                float* __restrict__ outp,
                                                unsigned short* __restrict__ outb) {
  const int row = blockIdx.x, t = threadIdx.x, g = t >> 3;
  const float mean = meanp[(size_t)row * Gc + g];
  const float rstd = rstdp[(size_t)row * Gc + g];
  for (int i = 0; i < 8; ++i) {
    int d = t * 8 + i;
    float v = (x[(size_t)row * Dc + d] - mean) * rstd * w[d] + bb[d];
    outp[(size_t)row * Dc + d] = v;
    outb[(size_t)row * Dc + d] = f32_to_bf16(v);
  }
}

// ---------------------------------------------------------------------------
// 4) complex EMA: one thread per (b,d), 8 complex states in registers
// ---------------------------------------------------------------------------
__global__ __launch_bounds__(256) void cema_kernel(const float* __restrict__ xin,
                                                   const float* __restrict__ alpha,
                                                   const float* __restrict__ delta,
                                                   const float* __restrict__ theta,
                                                   const float* __restrict__ gamma,
                                                   const float* __restrict__ omega,
                                                   float* __restrict__ outp) {
  const int t = blockIdx.x * blockDim.x + threadIdx.x;
  if (t >= Bc * Dc) return;
  const int b = t / Dc, d = t % Dc;
  const float th = sigmoidf_(theta[d]);
  const float om = omega[d];
  float ur[8], ui[8], qr[8], qi[8], gr[8], gi[8], hr[8], hi[8];
  for (int n = 0; n < 8; ++n) {
    float p  = sigmoidf_(alpha[d * 8 + n]);
    float qm = 1.f - p * sigmoidf_(delta[d * 8 + n]);
    float ph = th * (2.f * 3.14159265358979f * (float)(n + 1) / 8.f);
    float c = cosf(ph), sn = sinf(ph);
    ur[n] = p * c;  ui[n] = p * sn;
    qr[n] = qm * c; qi[n] = qm * sn;
    gr[n] = gamma[(d * 8 + n) * 2 + 0];
    gi[n] = gamma[(d * 8 + n) * 2 + 1];
    hr[n] = 0.f; hi[n] = 0.f;
  }
  const float* xp = xin + (size_t)b * Sc * Dc + d;
  float* op = outp + (size_t)b * Sc * Dc + d;
  for (int s = 0; s < Sc; ++s) {
    float xt = xp[(size_t)s * Dc];
    float y = 0.f;
    for (int n = 0; n < 8; ++n) {
      float nr = qr[n] * hr[n] - qi[n] * hi[n] + ur[n] * xt;
      float ni = qi[n] * hr[n] + qr[n] * hi[n] + ui[n] * xt;
      hr[n] = nr; hi[n] = ni;
      y += nr * gr[n] + ni * gi[n];
    }
    op[(size_t)s * Dc] = y + xt * om;
  }
}

// ---------------------------------------------------------------------------
// 5) row RMSNorm over D -> bf16 (GEMM A operand only)
// ---------------------------------------------------------------------------
__global__ __launch_bounds__(256) void rms_rows(const float* __restrict__ xin,
                                                const float* __restrict__ w,
                                                unsigned short* __restrict__ outb) {
  __shared__ float wsum[8];
  const int row = blockIdx.x, t = threadIdx.x;
  const float* x = xin + (size_t)row * Dc;
  float s = 0.f;
  for (int i = 0; i < 8; ++i) { float v = x[t * 8 + i]; s += v * v; }
  for (int m = 16; m >= 1; m >>= 1) s += __shfl_xor(s, m);
  if ((t & 31) == 0) wsum[t >> 5] = s;
  __syncthreads();
  float tot = 0.f;
  for (int i = 0; i < 8; ++i) tot += wsum[i];
  const float r = rsqrtf(tot / (float)Dc + 1e-5f);
  for (int i = 0; i < 8; ++i) {
    int d = t * 8 + i;
    outb[(size_t)row * Dc + d] = f32_to_bf16(x[d] * r * w[d]);
  }
}

// ---------------------------------------------------------------------------
// 6) bf16-WMMA GEMM: out = act(A @ W + bias + add)
//    A:  M x K bf16 row-major.   WT: N x K bf16 (pre-transposed weight).
//    Block tile 128x128, BK=32, 8 waves, double-buffered LDS, all B-frags
//    preloaded (partial dscnt waits), peeled last K-tile.
// ---------------------------------------------------------------------------
__global__ __launch_bounds__(256) void gemm_bf16(const unsigned short* __restrict__ A,
                                                 const unsigned short* __restrict__ WT,
                                                 const float* __restrict__ bias,
                                                 const float* __restrict__ addsrc,
                                                 float* __restrict__ outf,
                                                 unsigned short* __restrict__ outb,
                                                 unsigned short* __restrict__ outbT,
                                                 int M, int N, int K, int act) {
  __shared__ __align__(16) unsigned short As[2][128][40];  // [m][k]
  __shared__ __align__(16) unsigned short Bs[2][128][40];  // [n][k]
  const int tid = threadIdx.x, wave = tid >> 5, lane = tid & 31;
  const int lhi = lane >> 4, llo = lane & 15;
  const int bm = blockIdx.x * 128, bn = blockIdx.y * 128;
  const int wrow = wave * 16;
  const int sr = tid >> 1, sc0 = (tid & 1) * 16;   // staging: 16 bf16 / thread / tile

  v8f acc[8];
  for (int tn = 0; tn < 8; ++tn)
    for (int j = 0; j < 8; ++j) acc[tn][j] = 0.f;

  const unsigned short* aptr = A  + (size_t)(bm + sr) * K + sc0;
  const unsigned short* bptr = WT + (size_t)(bn + sr) * K + sc0;
  const int ntiles = K >> 5;

  auto stage = [&](int tt, int buf) {
    const unsigned short* an = aptr + tt * 32;
    const unsigned short* bp = bptr + tt * 32;
    *(u32x4*)&As[buf][sr][sc0]     = *(const u32x4*)(an);
    *(u32x4*)&As[buf][sr][sc0 + 8] = *(const u32x4*)(an + 8);
    *(u32x4*)&Bs[buf][sr][sc0]     = *(const u32x4*)(bp);
    *(u32x4*)&Bs[buf][sr][sc0 + 8] = *(const u32x4*)(bp + 8);
    const int tp = (tt + 1 < ntiles) ? tt + 1 : tt;    // branchless prefetch dist
    __builtin_prefetch(aptr + tp * 32, 0, 1);
    __builtin_prefetch(bptr + tp * 32, 0, 1);
  };

  auto compute = [&](int buf) {
    bfrag16 a;
    const int m = wrow + llo;
    a.q[0] = *(const u32x4*)&As[buf][m][lhi * 16];
    a.q[1] = *(const u32x4*)&As[buf][m][lhi * 16 + 8];
    bfrag16 bf[8];                       // all fragments live at once ->
#pragma unroll                           // 16 ds_load_b128 in flight, then
    for (int tn = 0; tn < 8; ++tn) {     // 8 WMMAs with partial waits
      const int n = tn * 16 + llo;
      bf[tn].q[0] = *(const u32x4*)&Bs[buf][n][lhi * 16];
      bf[tn].q[1] = *(const u32x4*)&Bs[buf][n][lhi * 16 + 8];
    }
#pragma unroll
    for (int tn = 0; tn < 8; ++tn)
      acc[tn] = __builtin_amdgcn_wmma_f32_16x16x32_bf16(
          false, a.v, false, bf[tn].v, (short)0, acc[tn], false, false);
  };

  stage(0, 0);
  __syncthreads();
  for (int t = 0; t < ntiles - 1; ++t) {   // branchless steady state
    stage(t + 1, (t + 1) & 1);
    compute(t & 1);
    __syncthreads();
  }
  compute((ntiles - 1) & 1);               // peeled last tile

  const int row0 = bm + wrow + lhi * 8;    // C layout: vgpr j -> M = j + 8*(lane>>4)
  if (outbT) {                             // transposed bf16: idx = (b*N+col)*S + s
    const int bq = row0 / Sc, s0 = row0 % Sc;
    for (int tn = 0; tn < 8; ++tn) {
      int col = bn + tn * 16 + llo;
      float bv = bias ? bias[col] : 0.f;
      pack8 pk;
      for (int j = 0; j < 8; ++j) {
        float v = acc[tn][j] + bv;
        if (act == 1) v = v * sigmoidf_(v);
        pk.u[j] = f32_to_bf16(v);
      }
      *(u32x4*)&outbT[((size_t)(bq * N) + col) * Sc + s0] = pk.q;
    }
  } else {
    for (int tn = 0; tn < 8; ++tn) {
      int col = bn + tn * 16 + llo;
      float bv = bias ? bias[col] : 0.f;
      for (int j = 0; j < 8; ++j) {
        int row = row0 + j;
        float v = acc[tn][j] + bv;
        if (addsrc) v += addsrc[(size_t)row * N + col];
        if (act == 1) v = v * sigmoidf_(v);
        if (outb) outb[(size_t)row * N + col] = f32_to_bf16(v);
        else      outf[(size_t)row * N + col] = v;
      }
    }
  }
}

// ---------------------------------------------------------------------------
// 7) head RMSNorm of z + q/k construction + rotary -> bf16 q,k
// ---------------------------------------------------------------------------
__global__ __launch_bounds__(512) void znorm_qk(const float* __restrict__ z,
                                                const float* __restrict__ ag,
                                                const float* __restrict__ ab,
                                                const float* __restrict__ freqs,
                                                unsigned short* __restrict__ qo,
                                                unsigned short* __restrict__ ko) {
  __shared__ float sq[512];
  __shared__ float qpre[512], kpre[512];
  const int row = blockIdx.x;           // b*S + s
  const int s = row % Sc;
  const int t = threadIdx.x;
  const int h = t >> 6, j = t & 63;
  const float v = z[(size_t)row * Zc + t];
  sq[t] = v * v;
  __syncthreads();
  for (int ofs = 32; ofs >= 1; ofs >>= 1) {
    if (j < ofs) sq[t] += sq[t + ofs];
    __syncthreads();
  }
  const float r  = rsqrtf(sq[h * 64] / 64.f + 1e-5f);
  const float zn = v * r;
  const float g0 = (ag[t] + 1.f) * 0.125f;          // includes 1/sqrt(ZH)
  const float g1 = (ag[Zc + t] + 1.f) * 0.125f;
  qpre[t] = zn * g0 + ab[t];
  kpre[t] = zn * g1 + ab[Zc + t];
  __syncthreads();
  const int pi = j >> 1;
  const float cs = freqs[((size_t)s * 32 + pi) * 2 + 0];
  const float sn = freqs[((size_t)s * 32 + pi) * 2 + 1];
  float qv, kv;
  if ((j & 1) == 0) { qv = qpre[t] * cs - qpre[t + 1] * sn;
                      kv = kpre[t] * cs - kpre[t + 1] * sn; }
  else              { qv = qpre[t - 1] * sn + qpre[t] * cs;
                      kv = kpre[t - 1] * sn + kpre[t] * cs; }
  qo[(size_t)row * Zc + t] = f32_to_bf16(qv);
  ko[(size_t)row * Zc + t] = f32_to_bf16(kv);
}

// ---------------------------------------------------------------------------
// 8) causal flash attention. Block = (32 q-rows, h, b), 8 waves.
//    Q/K bf16 row-major; V bf16 pre-transposed (b, col, s) -> LDS copy only.
//    P written bf16 by softmax; all 8 V fragments preloaded in P.V chain.
//    Epilogue fuses r-gate, writes bf16.
// ---------------------------------------------------------------------------
__global__ __launch_bounds__(256) void attn_flash(const unsigned short* __restrict__ Q,
                                                  const unsigned short* __restrict__ Km,
                                                  const unsigned short* __restrict__ VT,
                                                  const float* __restrict__ Rg,
                                                  unsigned short* __restrict__ Ob) {
  __shared__ __align__(16) unsigned short Qs[32][72];    // [q][feat]
  __shared__ __align__(16) unsigned short Ks[32][72];    // [key][feat]
  __shared__ __align__(16) unsigned short Vt[512][40];   // [vcol][key]
  __shared__ __align__(16) unsigned short Pb[32][40];    // bf16 probabilities
  __shared__ float Ps[32][36];
  __shared__ float mrow[32], lrow[32], srow[32];
  const int tid = threadIdx.x, wave = tid >> 5, lane = tid & 31;
  const int lhi = lane >> 4, llo = lane & 15;
  const int qb = blockIdx.x, h = blockIdx.y, b = blockIdx.z;
  const int q0 = qb * 32;

  {   // stage Q: 32x64 bf16 copy (16B per thread)
    int r = tid >> 3, c0 = (tid & 7) * 8;
    const unsigned short* src = Q + (size_t)(b * Sc + q0 + r) * Zc + h * 64 + c0;
    *(u32x4*)&Qs[r][c0] = *(const u32x4*)src;
  }
  if (tid < 32) { mrow[tid] = -3.0e38f; lrow[tid] = 0.f; srow[tid] = 1.f; }

  v8f acc[8];
  for (int t = 0; t < 8; ++t)
    for (int j = 0; j < 8; ++j) acc[t][j] = 0.f;
  const int rstrip = (wave >> 2) * 16;       // 0 or 16
  const int cbase  = (wave & 3) * 128;       // 0..384
  const int nKb = qb + 1;                    // causal

  for (int jb = 0; jb < nKb; ++jb) {
    const int k0 = jb * 32;
    __syncthreads();                         // protect LDS reuse
    {   // stage K: 32x64 bf16 copy
      int r = tid >> 3, c0 = (tid & 7) * 8;
      const unsigned short* src = Km + (size_t)(b * Sc + k0 + r) * Zc + h * 64 + c0;
      *(u32x4*)&Ks[r][c0] = *(const u32x4*)src;
    }
    {   // stage V: pre-transposed global -> pure 16B copies (2 vcols/thread)
      for (int cc = 0; cc < 2; ++cc) {
        int c = tid + cc * 256;
        const unsigned short* src = VT + ((size_t)(b * HDc) + h * 512 + c) * Sc + k0;
        *(u32x4*)&Vt[c][0]  = *(const u32x4*)(src);
        *(u32x4*)&Vt[c][8]  = *(const u32x4*)(src + 8);
        *(u32x4*)&Vt[c][16] = *(const u32x4*)(src + 16);
        *(u32x4*)&Vt[c][24] = *(const u32x4*)(src + 24);
      }
    }
    __syncthreads();

    // S = Q Kt (32x32): 2 WMMA k-steps over 64 features; waves 0-3
    if (wave < 4) {
      const int rs = (wave >> 1) * 16, cs = (wave & 1) * 16;
      v8f sacc;
      for (int j = 0; j < 8; ++j) sacc[j] = 0.f;
      for (int ks = 0; ks < 64; ks += 32) {
        bfrag16 a, bb;
        int m = rs + llo;
        a.q[0] = *(const u32x4*)&Qs[m][ks + lhi * 16];
        a.q[1] = *(const u32x4*)&Qs[m][ks + lhi * 16 + 8];
        int n = cs + llo;
        bb.q[0] = *(const u32x4*)&Ks[n][ks + lhi * 16];
        bb.q[1] = *(const u32x4*)&Ks[n][ks + lhi * 16 + 8];
        sacc = __builtin_amdgcn_wmma_f32_16x16x32_bf16(
            false, a.v, false, bb.v, (short)0, sacc, false, false);
      }
      int n = cs + llo, kc = k0 + n;
      for (int j = 0; j < 8; ++j) {
        int r = rs + lhi * 8 + j;
        float sv = sacc[j];
        if (kc > q0 + r) sv = -3.0e38f;      // causal mask
        Ps[r][n] = sv;
      }
    }
    __syncthreads();

    // online softmax; writes bf16 P
    if (tid < 32) {
      float mo = mrow[tid], mn = mo;
      for (int j = 0; j < 32; ++j) mn = fmaxf(mn, Ps[tid][j]);
      float sc = __expf(mo - mn);
      float ls = lrow[tid] * sc;
      for (int j = 0; j < 32; ++j) {
        float p = __expf(Ps[tid][j] - mn);
        Pb[tid][j] = f32_to_bf16(p);
        ls += p;
      }
      mrow[tid] = mn; lrow[tid] = ls; srow[tid] = sc;
    }
    __syncthreads();

    // rescale accumulators; acc += P (32x32) x V (32x512)
    float sc8[8];
    for (int j = 0; j < 8; ++j) sc8[j] = srow[rstrip + lhi * 8 + j];
    for (int t = 0; t < 8; ++t)
      for (int j = 0; j < 8; ++j) acc[t][j] *= sc8[j];
    bfrag16 a;
    {
      int m = rstrip + llo;
      a.q[0] = *(const u32x4*)&Pb[m][lhi * 16];
      a.q[1] = *(const u32x4*)&Pb[m][lhi * 16 + 8];
    }
    bfrag16 bf[8];                           // preload all 8 V fragments
#pragma unroll
    for (int t = 0; t < 8; ++t) {
      const int n = cbase + t * 16 + llo;
      bf[t].q[0] = *(const u32x4*)&Vt[n][lhi * 16];
      bf[t].q[1] = *(const u32x4*)&Vt[n][lhi * 16 + 8];
    }
#pragma unroll
    for (int t = 0; t < 8; ++t)
      acc[t] = __builtin_amdgcn_wmma_f32_16x16x32_bf16(
          false, a.v, false, bf[t].v, (short)0, acc[t], false, false);
  }

  // epilogue: normalize, fuse r-gate, write bf16
  float li[8];
  for (int j = 0; j < 8; ++j) li[j] = 1.f / lrow[rstrip + lhi * 8 + j];
  for (int t = 0; t < 8; ++t) {
    int n = cbase + t * 16 + llo;
    for (int j = 0; j < 8; ++j) {
      int r = rstrip + lhi * 8 + j;
      size_t idx = (size_t)(b * Sc + q0 + r) * HDc + h * 512 + n;
      Ob[idx] = f32_to_bf16(acc[t][j] * li[j] * Rg[idx]);
    }
  }
}

// ---------------------------------------------------------------------------
// launch
// ---------------------------------------------------------------------------
extern "C" void kernel_launch(void* const* d_in, const int* in_sizes, int n_in,
                              void* d_out, int out_size, void* d_ws, size_t ws_size,
                              hipStream_t stream) {
  (void)in_sizes; (void)n_in; (void)out_size; (void)ws_size;
  const float* x      = (const float*)d_in[0];
  const float* tnw    = (const float*)d_in[1];
  const float* tnb    = (const float*)d_in[2];
  const float* alpha  = (const float*)d_in[3];
  const float* delta  = (const float*)d_in[4];
  const float* theta  = (const float*)d_in[5];
  const float* egam   = (const float*)d_in[6];
  const float* omega  = (const float*)d_in[7];
  const float* rmsw   = (const float*)d_in[8];
  const float* wzw    = (const float*)d_in[9];
  const float* wzb    = (const float*)d_in[10];
  const float* wvw    = (const float*)d_in[11];
  const float* wvb    = (const float*)d_in[12];
  const float* wrw    = (const float*)d_in[13];
  const float* wrb    = (const float*)d_in[14];
  const float* wh1w   = (const float*)d_in[15];
  const float* wh1b   = (const float*)d_in[16];
  const float* wh2w   = (const float*)d_in[17];
  const float* agam   = (const float*)d_in[18];
  const float* abet   = (const float*)d_in[19];
  const float* freqs  = (const float*)d_in[20];

  const size_t BSG = (size_t)Bc * Sc * Gc;
  const size_t BSD = (size_t)Bc * Sc * Dc;
  const size_t BSZ = (size_t)Bc * Sc * Zc;
  const size_t BSH = (size_t)Bc * Sc * HDc;

  char* p = (char*)d_ws;
  auto alloc_f = [&](size_t n) { float* r = (float*)p;
                                 p += ((n * sizeof(float) + 255) & ~(size_t)255); return r; };
  auto alloc_h = [&](size_t n) { unsigned short* r = (unsigned short*)p;
                                 p += ((n * 2 + 255) & ~(size_t)255); return r; };

  float* s1p   = alloc_f(BSG);
  float* s2p   = alloc_f(BSG);
  float* meanp = alloc_f(BSG);
  float* rstdp = alloc_f(BSG);
  float* tsn   = alloc_f(BSD);
  float* cem   = alloc_f(BSD);          // reused as tmp1 after mx is built
  float* zp    = alloc_f(BSZ);
  float* rp    = alloc_f(BSH);
  unsigned short* tsnb  = alloc_h(BSD);
  unsigned short* mxb   = alloc_h(BSD);
  unsigned short* qb16  = alloc_h(BSZ);
  unsigned short* kb16  = alloc_h(BSZ);
  unsigned short* vT16  = alloc_h(BSH);   // (b, col, s) transposed
  unsigned short* ab16  = alloc_h(BSH);
  unsigned short* wzT  = alloc_h((size_t)Dc * Zc);    // Z  x D
  unsigned short* wvT  = alloc_h((size_t)Dc * HDc);   // HD x D
  unsigned short* wrT  = alloc_h((size_t)Dc * HDc);   // HD x D
  unsigned short* wh1T = alloc_h((size_t)Dc * Dc);    // D  x D
  unsigned short* wh2T = alloc_h((size_t)HDc * Dc);   // D  x HD
  float* tmp1 = cem;

  const int M = Bc * Sc;   // 4096
  const dim3 tb(32, 8);

  // weights -> bf16 transposed (N x K); resident in 192MB L2 afterwards
  cvt_t<<<dim3(Zc  / 32, Dc  / 32), tb, 0, stream>>>(wzw,  wzT,  Dc,  Zc);
  cvt_t<<<dim3(HDc / 32, Dc  / 32), tb, 0, stream>>>(wvw,  wvT,  Dc,  HDc);
  cvt_t<<<dim3(HDc / 32, Dc  / 32), tb, 0, stream>>>(wrw,  wrT,  Dc,  HDc);
  cvt_t<<<dim3(Dc  / 32, Dc  / 32), tb, 0, stream>>>(wh1w, wh1T, Dc,  Dc);
  cvt_t<<<dim3(Dc  / 32, HDc / 32), tb, 0, stream>>>(wh2w, wh2T, HDc, Dc);

  tsn_partial<<<M, 256, 0, stream>>>(x, s1p, s2p);
  tsn_scan<<<1, 64, 0, stream>>>(s1p, s2p, meanp, rstdp);
  tsn_norm<<<M, 256, 0, stream>>>(x, meanp, rstdp, tnw, tnb, tsn, tsnb);
  cema_kernel<<<(Bc * Dc) / 256, 256, 0, stream>>>(tsn, alpha, delta, theta, egam, omega, cem);
  rms_rows<<<M, 256, 0, stream>>>(cem, rmsw, mxb);

  gemm_bf16<<<dim3(M / 128, Zc / 128), 256, 0, stream>>>(
      mxb, wzT, wzb, nullptr, zp, nullptr, nullptr, M, Zc, Dc, 0);
  znorm_qk<<<M, 512, 0, stream>>>(zp, agam, abet, freqs, qb16, kb16);
  gemm_bf16<<<dim3(M / 128, HDc / 128), 256, 0, stream>>>(
      tsnb, wvT, wvb, nullptr, nullptr, nullptr, vT16, M, HDc, Dc, 1); // v: silu -> bf16^T
  gemm_bf16<<<dim3(M / 128, HDc / 128), 256, 0, stream>>>(
      mxb, wrT, wrb, nullptr, rp, nullptr, nullptr, M, HDc, Dc, 1);    // r: silu -> f32

  attn_flash<<<dim3(Sc / 32, Hc, Bc), 256, 0, stream>>>(qb16, kb16, vT16, rp, ab16);

  gemm_bf16<<<dim3(M / 128, Dc / 128), 256, 0, stream>>>(
      mxb, wh1T, wh1b, x, tmp1, nullptr, nullptr, M, Dc, Dc, 0);       // mx@wh1+b+residual
  gemm_bf16<<<dim3(M / 128, Dc / 128), 256, 0, stream>>>(
      ab16, wh2T, nullptr, tmp1, (float*)d_out, nullptr, nullptr, M, Dc, HDc, 0);
}
